// PathFormerModel_73134703116730
// MI455X (gfx1250) — matrix operand order repaired and checked by reading, not compiled, not run on hardware
//
#include <hip/hip_runtime.h>

#define B_    8
#define L_    96
#define N_    321
#define D_    32
#define PRED_ 96
#define DFF_  128
#define E_    4
#define SZ_   (B_*L_*N_*D_)   /* 7,888,896 */
#define BLN_  (B_*L_*N_)      /* 246,528   */
#define EPS_  1e-5f

typedef __attribute__((ext_vector_type(16))) _Float16 v16h;
typedef __attribute__((ext_vector_type(8)))  float    v8f;

// ---------------------------------------------------------------------------
// Generic GEMM: C[M,N] = A[M,K] @ W[K,N] + bias + pos[row%pnmod, :]
// fp32 in/out, f16 WMMA compute. One 16x16 C tile per wave.
// N % 16 == 0, K % 32 == 0; M may be ragged.
// Ragged-M handling: A-row index is CLAMPED (not predicated) — C row m depends
// only on A row m, and out-of-range C rows are never stored, so clamped rows
// are harmless. This keeps every load unconditional (no exec-mask churn).
// ---------------------------------------------------------------------------
__global__ void gemm_wmma_f16(const float* __restrict__ A, const float* __restrict__ W,
                              const float* __restrict__ bias, const float* __restrict__ pos,
                              int pnmod, float* __restrict__ C, int M, int K, int N)
{
    int ntn  = N >> 4;
    int ntm  = (M + 15) >> 4;
    int tile = blockIdx.x * (blockDim.x >> 5) + (threadIdx.x >> 5);
    if (tile >= ntm * ntn) return;          // uniform per wave: EXEC stays all-1s
    int tm = tile / ntn, tn = tile % ntn;
    int lane = threadIdx.x & 31;
    int half = lane >> 4;                   // lane group 0-15 / 16-31
    int l16  = lane & 15;

    int arow = tm * 16 + l16;
    if (arow >= M) arow = M - 1;            // clamp (see note above)
    const float2* Arow = (const float2*)(A + (size_t)arow * K);
    int bcol = tn * 16 + l16;

    v8f acc = {};
    for (int kb = 0; kb < K; kb += 32) {
        v16h af, bf;
        // A 16x32 f16 fragment: lanes 0-15 -> K {0..7,16..23}, lanes 16-31 -> K {8..15,24..31}
#pragma unroll
        for (int v = 0; v < 8; ++v) {
            int k0 = kb + ((v & 4) ? 16 : 0) + half * 8 + 2 * (v & 3);  // always even
            float2 x = Arow[k0 >> 1];
            af[2 * v]     = (_Float16)x.x;
            af[2 * v + 1] = (_Float16)x.y;
        }
        // B 32x16 f16 fragment: lanes 0-15 -> K {0..7,16..23}, lanes 16-31 -> K {8..15,24..31}
#pragma unroll
        for (int v = 0; v < 8; ++v) {
            int k = kb + ((v & 4) ? 16 : 0) + half * 8 + 2 * (v & 3);
            bf[2 * v]     = (_Float16)W[(size_t)k * N + bcol];
            bf[2 * v + 1] = (_Float16)W[(size_t)(k + 1) * N + bcol];
        }
        acc = __builtin_amdgcn_wmma_f32_16x16x32_f16(false, af, false, bf,
                                                     (short)0, acc, false, false);
    }
    float badd = bias ? bias[bcol] : 0.f;
#pragma unroll
    for (int v = 0; v < 8; ++v) {
        int row = tm * 16 + v + half * 8;   // C layout: VGPR v -> rows v / v+8
        if (row < M) {
            float o = acc[v] + badd;
            if (pos) o += pos[(row % pnmod) * N + bcol];
            C[(size_t)row * N + bcol] = o;
        }
    }
}

// ---------------------------------------------------------------------------
// RevIN stats + normalize + start_fc:  out4[b,l,n,d] = xn*w[d]+b[d]
// ---------------------------------------------------------------------------
__global__ void revin_start(const float* __restrict__ x, const float* __restrict__ w,
                            const float* __restrict__ bi, float* __restrict__ mean,
                            float* __restrict__ stdev, float* __restrict__ out4)
{
    int bn = blockIdx.x; int b = bn / N_, n = bn % N_;
    int t = threadIdx.x;                       // 96 threads = L
    __shared__ float sx[L_];
    __shared__ float stats[2];
    float xv = x[((size_t)b * L_ + t) * N_ + n];
    sx[t] = xv;
    __syncthreads();
    if (t == 0) {
        float m = 0.f;
        for (int i = 0; i < L_; ++i) m += sx[i];
        m /= (float)L_;
        float v = 0.f;
        for (int i = 0; i < L_; ++i) { float d = sx[i] - m; v += d * d; }
        v /= (float)L_;
        stats[0] = m; stats[1] = sqrtf(v + EPS_);
        mean[bn] = m; stdev[bn] = stats[1];
    }
    __syncthreads();
    float xn = (xv - stats[0]) / stats[1];
    size_t o = (((size_t)b * L_ + t) * N_ + n) * D_;
#pragma unroll
    for (int d = 0; d < D_; ++d) out4[o + d] = xn * w[d] + bi[d];
}

// ---------------------------------------------------------------------------
// Season (47-bin DFT, top-3 reconstruct) + multi-kernel trend, per (b,n)
// ---------------------------------------------------------------------------
__global__ void season_trend(const float* __restrict__ x4, const float* __restrict__ dw,
                             const float* __restrict__ db, float* __restrict__ s_out)
{
    int bn = blockIdx.x; int b = bn / N_, n = bn % N_;
    int t = threadIdx.x;                       // 96 threads
    __shared__ float sx[L_];
    __shared__ float Fr[48], Fi[48], mag[48];
    __shared__ int topi[3];
    float xv = x4[(((size_t)b * L_ + t) * N_ + n) * D_];   // d = 0 slice
    sx[t] = xv;
    __syncthreads();
    if (t >= 1 && t <= 47) {
        float fr = 0.f, fi = 0.f;
        float w = 6.2831853071795864769f * (float)t / 96.f;
        for (int tt = 0; tt < 96; ++tt) {
            float th = w * (float)tt;
            fr += sx[tt] * cosf(th);
            fi -= sx[tt] * sinf(th);
        }
        Fr[t] = fr; Fi[t] = fi; mag[t] = fr * fr + fi * fi;
    }
    __syncthreads();
    if (t == 0) {   // top-3 magnitudes (ties -> lowest index, like lax.top_k)
        int c0 = 1; float best = mag[1];
        for (int k = 2; k < 48; ++k) if (mag[k] > best) { best = mag[k]; c0 = k; }
        int c1 = -1; best = -1e30f;
        for (int k = 1; k < 48; ++k) if (k != c0 && mag[k] > best) { best = mag[k]; c1 = k; }
        int c2 = -1; best = -1e30f;
        for (int k = 1; k < 48; ++k) if (k != c0 && k != c1 && mag[k] > best) { best = mag[k]; c2 = k; }
        topi[0] = c0; topi[1] = c1; topi[2] = c2;
    }
    __syncthreads();
    float wave = 0.f;
#pragma unroll
    for (int j = 0; j < 3; ++j) {
        int k = topi[j];
        float w  = 6.2831853071795864769f * (float)k / 96.f;
        float th = w * (float)t;
        wave += (2.f / 96.f) * (Fr[k] * cosf(th) - Fi[k] * sinf(th));
    }
    const int kern[3] = {4, 8, 12};
    float means[3];
#pragma unroll
    for (int j = 0; j < 3; ++j) {
        int k = kern[j], f = (k - 1) / 2;
        float a = 0.f;
        for (int o = -f; o <= k - 1 - f; ++o) {
            int ti = t + o; ti = ti < 0 ? 0 : (ti > 95 ? 95 : ti);
            a += sx[ti];
        }
        means[j] = a / (float)k;
    }
    float l0 = xv * dw[0] + db[0], l1 = xv * dw[1] + db[1], l2 = xv * dw[2] + db[2];
    float mx = fmaxf(l0, fmaxf(l1, l2));
    float e0 = expf(l0 - mx), e1 = expf(l1 - mx), e2 = expf(l2 - mx);
    float trend = (means[0] * e0 + means[1] * e1 + means[2] * e2) / (e0 + e1 + e2);
    s_out[((size_t)b * L_ + t) * N_ + n] = xv + wave + trend;
}

// ---------------------------------------------------------------------------
// Gating: g = s@slw+slb; logits = g@w_gate; top-2 softmax scatter; cv^2 aux
// ---------------------------------------------------------------------------
__global__ void gate_kernel(const float* __restrict__ s, const float* __restrict__ slw,
                            const float* __restrict__ slb, const float* __restrict__ wg,
                            float* __restrict__ gates, float* __restrict__ balance)
{
    __shared__ float g[B_ * L_];
    __shared__ float logits[B_ * E_];
    __shared__ float gl[B_ * E_];
    int tid = threadIdx.x;
    for (int i = tid; i < B_ * L_; i += blockDim.x) {
        int b = i / L_, t = i % L_;
        float a = slb[0];
        const float* row = s + ((size_t)b * L_ + t) * N_;
        for (int n = 0; n < N_; ++n) a += row[n] * slw[n];
        g[i] = a;
    }
    __syncthreads();
    if (tid < B_ * E_) {
        int b = tid / E_, e = tid % E_;
        float a = 0.f;
        for (int t = 0; t < L_; ++t) a += g[b * L_ + t] * wg[t * E_ + e];
        logits[tid] = a;
        gl[tid] = 0.f;
    }
    __syncthreads();
    if (tid < B_) {
        int b = tid, i0 = -1, i1 = -1;
        float v0 = -1e30f, v1 = -1e30f;
        for (int e = 0; e < E_; ++e) {
            float v = logits[b * E_ + e];
            if (v > v0)      { v1 = v0; i1 = i0; v0 = v; i0 = e; }
            else if (v > v1) { v1 = v;  i1 = e; }
        }
        float e0 = expf(v0 - v0), e1 = expf(v1 - v0);
        float inv = 1.f / (e0 + e1);
        gl[b * E_ + i0] = e0 * inv;
        gl[b * E_ + i1] = e1 * inv;
    }
    __syncthreads();
    if (tid == 0) {
        float imp[E_], ld[E_];
        for (int e = 0; e < E_; ++e) {
            float si = 0.f, sl = 0.f;
            for (int b = 0; b < B_; ++b) {
                float gv = gl[b * E_ + e];
                si += gv; sl += (gv > 0.f) ? 1.f : 0.f;
            }
            imp[e] = si; ld[e] = sl;
        }
        float aux = 0.f;
        for (int pass = 0; pass < 2; ++pass) {
            const float* v = pass ? ld : imp;
            float m = 0.f;
            for (int e = 0; e < E_; ++e) m += v[e];
            m /= (float)E_;
            float var = 0.f;
            for (int e = 0; e < E_; ++e) { float d = v[e] - m; var += d * d; }
            var /= (float)(E_ - 1);                   // ddof=1
            aux += var / (m * m + 1e-10f);
        }
        balance[0] += 1e-2f * aux;
    }
    if (tid < B_ * E_) gates[tid] = gl[tid];
}

// ---------------------------------------------------------------------------
// Patchify: xr[(b*N+n)*pn + p, dd*P+pp] = x4[b, p*P+pp, n, dd]
// ---------------------------------------------------------------------------
__global__ void patchify(const float* __restrict__ x, float* __restrict__ xr, int P, int pn)
{
    int idx = blockIdx.x * blockDim.x + threadIdx.x;   // over SZ_ (grid exact)
    int iD = D_ * P;
    int f = idx % iD;  int r  = idx / iD;
    int p = r % pn;    int bn = r / pn;
    int n = bn % N_;   int b  = bn / N_;
    int dd = f / P;    int pp = f % P;
    xr[idx] = x[(((size_t)b * L_ + p * P + pp) * N_ + n) * D_ + dd];
}

// ---------------------------------------------------------------------------
// Attention per (seq, head): pn*dk == 96 always. One wave per head instance.
// ---------------------------------------------------------------------------
__global__ void attn_kernel(const float* __restrict__ Q, const float* __restrict__ Km,
                            const float* __restrict__ V, float* __restrict__ O,
                            int pn, int dk)
{
    __shared__ float sq[4][96], sk[4][96], sv[4][96], ss[4][24 * 24];
    int w = threadIdx.x >> 5, lane = threadIdx.x & 31;
    int inst = blockIdx.x * 4 + w;                 // grid exact: 2568*32 / 4
    int seq = inst >> 5, head = inst & 31;
    int iD = D_ * dk;
    size_t base = (size_t)seq * 3072 + head * dk;  // pn*iD == 3072
    for (int i = lane; i < 96; i += 32) {
        int p = i / dk, kk = i % dk;
        size_t a = base + (size_t)p * iD + kk;
        sq[w][i] = Q[a]; sk[w][i] = Km[a]; sv[w][i] = V[a];
    }
    __syncthreads();
    float scale = rsqrtf((float)dk);
    for (int i = lane; i < pn * pn; i += 32) {
        int qi = i / pn, kj = i % pn;
        float s = 0.f;
        for (int kk = 0; kk < dk; ++kk) s += sq[w][qi * dk + kk] * sk[w][kj * dk + kk];
        ss[w][i] = s * scale;
    }
    __syncthreads();
    for (int qi = lane; qi < pn; qi += 32) {
        float mx = -1e30f;
        for (int kj = 0; kj < pn; ++kj) mx = fmaxf(mx, ss[w][qi * pn + kj]);
        float sum = 0.f;
        for (int kj = 0; kj < pn; ++kj) {
            float e = expf(ss[w][qi * pn + kj] - mx);
            ss[w][qi * pn + kj] = e; sum += e;
        }
        float inv = 1.f / sum;
        for (int kj = 0; kj < pn; ++kj) ss[w][qi * pn + kj] *= inv;
    }
    __syncthreads();
    for (int i = lane; i < 96; i += 32) {
        int p = i / dk, kk = i % dk;
        float o = 0.f;
        for (int kj = 0; kj < pn; ++kj) o += ss[w][p * pn + kj] * sv[w][kj * dk + kk];
        O[base + (size_t)p * iD + kk] = o;
    }
}

// ---------------------------------------------------------------------------
// y1 = LayerNorm(x + o) over last-dim D (the raw reshape is a no-op in memory)
// ---------------------------------------------------------------------------
__global__ void ln_res(const float* __restrict__ x, const float* __restrict__ o,
                       const float* __restrict__ g, const float* __restrict__ bta,
                       float* __restrict__ y)
{
    int tok = blockIdx.x * blockDim.x + threadIdx.x;   // over BLN_ (grid exact)
    float v[D_];
    float m = 0.f;
#pragma unroll
    for (int d = 0; d < D_; ++d) {
        float t = x[(size_t)tok * D_ + d] + o[(size_t)tok * D_ + d];
        v[d] = t; m += t;
    }
    m *= (1.f / (float)D_);
    float var = 0.f;
#pragma unroll
    for (int d = 0; d < D_; ++d) { float dd = v[d] - m; var += dd * dd; }
    var *= (1.f / (float)D_);
    float r = rsqrtf(var + EPS_);
#pragma unroll
    for (int d = 0; d < D_; ++d)
        y[(size_t)tok * D_ + d] = (v[d] - m) * r * g[d] + bta[d];
}

// ---------------------------------------------------------------------------
// Fused FFN (32->128 GELU ->32) + residual + LN2 + gated accumulate
// ---------------------------------------------------------------------------
__global__ void ffn_ln2_acc(const float* __restrict__ y1,
                            const float* __restrict__ w1, const float* __restrict__ b1,
                            const float* __restrict__ w2, const float* __restrict__ b2,
                            const float* __restrict__ g2, const float* __restrict__ bb2,
                            const float* __restrict__ gates, int expert,
                            float* __restrict__ acc)
{
    __shared__ float s1[D_ * DFF_], s2[DFF_ * D_], sb1[DFF_], sb2[D_], sg[D_], sbb[D_];
    for (int i = threadIdx.x; i < D_ * DFF_; i += blockDim.x) { s1[i] = w1[i]; s2[i] = w2[i]; }
    for (int i = threadIdx.x; i < DFF_; i += blockDim.x) sb1[i] = b1[i];
    for (int i = threadIdx.x; i < D_; i += blockDim.x) { sb2[i] = b2[i]; sg[i] = g2[i]; sbb[i] = bb2[i]; }
    __syncthreads();
    int tok = blockIdx.x * blockDim.x + threadIdx.x;   // over BLN_ (grid exact)
    float yv[D_], o2[D_];
#pragma unroll
    for (int d = 0; d < D_; ++d) { yv[d] = y1[(size_t)tok * D_ + d]; o2[d] = sb2[d]; }
    for (int j = 0; j < DFF_; ++j) {
        float h = sb1[j];
#pragma unroll
        for (int d = 0; d < D_; ++d) h += yv[d] * s1[d * DFF_ + j];
        h = 0.5f * h * (1.f + erff(h * 0.70710678118654752f));   // exact GELU
#pragma unroll
        for (int d = 0; d < D_; ++d) o2[d] += h * s2[j * D_ + d];
    }
    float m = 0.f;
#pragma unroll
    for (int d = 0; d < D_; ++d) { o2[d] += yv[d]; m += o2[d]; }
    m *= (1.f / (float)D_);
    float var = 0.f;
#pragma unroll
    for (int d = 0; d < D_; ++d) { float t = o2[d] - m; var += t * t; }
    var *= (1.f / (float)D_);
    float r = rsqrtf(var + EPS_);
    float gate = gates[(tok / (L_ * N_)) * E_ + expert];
#pragma unroll
    for (int d = 0; d < D_; ++d)
        acc[(size_t)tok * D_ + d] += gate * ((o2[d] - m) * r * sg[d] + sbb[d]);
}

// ---------------------------------------------------------------------------
// Misc small kernels
// ---------------------------------------------------------------------------
__global__ void copy4(float4* __restrict__ dst, const float4* __restrict__ src)
{
    size_t i = (size_t)blockIdx.x * blockDim.x + threadIdx.x;
    dst[i] = src[i];
}

__global__ void mk_h(const float* __restrict__ x, float* __restrict__ h)
{
    int idx = blockIdx.x * blockDim.x + threadIdx.x;   // over SZ_
    int d = idx % D_;  int r  = idx / D_;
    int l = r % L_;    int r2 = r / L_;
    int n = r2 % N_;   int b  = r2 / N_;
    h[idx] = x[(((size_t)b * L_ + l) * N_ + n) * D_ + d];
}

__global__ void zero_scalar(float* p) { p[0] = 0.f; }

__global__ void finalize(const float* __restrict__ y, const float* __restrict__ mean,
                         const float* __restrict__ stdev, const float* __restrict__ bal,
                         float* __restrict__ out)
{
    int idx = blockIdx.x * blockDim.x + threadIdx.x;   // over B*PRED*N (grid exact)
    if (idx == 0) out[B_ * PRED_ * N_] = bal[0];
    int n = idx % N_; int r = idx / N_;
    int pred = r % PRED_; int b = r / PRED_;
    out[idx] = y[((size_t)b * N_ + n) * PRED_ + pred] * stdev[b * N_ + n] + mean[b * N_ + n];
}

// ---------------------------------------------------------------------------
// Host side
// ---------------------------------------------------------------------------
static void launch_gemm(const float* A, const float* W, const float* bias, const float* pos,
                        int pnmod, float* C, int M, int K, int N, hipStream_t s)
{
    int ntm = (M + 15) / 16, ntn = N / 16;
    long tiles = (long)ntm * ntn;
    int blocks = (int)((tiles + 7) / 8);   // 8 waves (256 threads) per block
    gemm_wmma_f16<<<blocks, 256, 0, s>>>(A, W, bias, pos, pnmod, C, M, K, N);
}

extern "C" void kernel_launch(void* const* d_in, const int* in_sizes, int n_in,
                              void* d_out, int out_size, void* d_ws, size_t ws_size,
                              hipStream_t stream)
{
    (void)in_sizes; (void)n_in; (void)out_size; (void)ws_size;
    auto P_ = [&](int i) { return (const float*)d_in[i]; };
    // Input layout (depth-first insertion order):
    // 0:x 1:start_fc.w 2:start_fc.b
    // per layer l (base 3+81*l): +0 slw +1 slb +2 w_gate +3 decomp_w +4 decomp_b
    //   per expert e (base+5+19*e): emb_w,emb_b,pos,wq,bq,wk,bk,wv,bv,wo,bo,
    //                               ln1_g,ln1_b,ff1_w,ff1_b,ff2_w,ff2_b,ln2_g,ln2_b
    // 165: proj_w  166: proj_b
    const float* xin = P_(0);

    float* wsf   = (float*)d_ws;
    float* bufA  = wsf;
    float* bufB  = bufA + SZ_;
    float* bufP  = bufB + SZ_;
    float* bufZ  = bufP + SZ_;
    float* bufK  = bufZ + SZ_;
    float* bufV  = bufK + SZ_;
    float* sbuf  = bufV + SZ_;          // BLN_ floats, reused as proj output
    float* meanb = sbuf + BLN_;         // B*N
    float* stdb  = meanb + B_ * N_;     // B*N
    float* gatesb = stdb + B_ * N_;     // B*E
    float* balb  = gatesb + B_ * E_;    // 1

    zero_scalar<<<1, 1, 0, stream>>>(balb);
    revin_start<<<B_ * N_, L_, 0, stream>>>(xin, P_(1), P_(2), meanb, stdb, bufA);

    const int PATCH[4] = {16, 12, 8, 4};
    for (int l = 0; l < 2; ++l) {
        float* xcur = (l == 0) ? bufA : bufB;
        float* accb = (l == 0) ? bufB : bufA;
        int LB = 3 + 81 * l;

        copy4<<<SZ_ / 1024, 256, 0, stream>>>((float4*)accb, (const float4*)xcur);
        season_trend<<<B_ * N_, L_, 0, stream>>>(xcur, P_(LB + 3), P_(LB + 4), sbuf);
        gate_kernel<<<1, 128, 0, stream>>>(sbuf, P_(LB), P_(LB + 1), P_(LB + 2), gatesb, balb);

        for (int e = 0; e < 4; ++e) {
            int EB = LB + 5 + 19 * e;
            int Pp = PATCH[e], pn = L_ / Pp, iD = D_ * Pp, M = B_ * N_ * pn;
            patchify<<<SZ_ / 256, 256, 0, stream>>>(xcur, bufP, Pp, pn);
            // z = xr@emb_w + emb_b + pos
            launch_gemm(bufP, P_(EB + 0), P_(EB + 1), P_(EB + 2), pn, bufZ, M, iD, iD, stream);
            // q,k,v
            launch_gemm(bufZ, P_(EB + 3), P_(EB + 4), nullptr, 1, bufP, M, iD, iD, stream);
            launch_gemm(bufZ, P_(EB + 5), P_(EB + 6), nullptr, 1, bufK, M, iD, iD, stream);
            launch_gemm(bufZ, P_(EB + 7), P_(EB + 8), nullptr, 1, bufV, M, iD, iD, stream);
            attn_kernel<<<(B_ * N_ * D_) / 4, 128, 0, stream>>>(bufP, bufK, bufV, bufZ, pn, Pp);
            // o = ctx@wo + bo
            launch_gemm(bufZ, P_(EB + 9), P_(EB + 10), nullptr, 1, bufP, M, iD, iD, stream);
            ln_res<<<BLN_ / 256, 256, 0, stream>>>(xcur, bufP, P_(EB + 11), P_(EB + 12), bufK);
            ffn_ln2_acc<<<BLN_ / 256, 256, 0, stream>>>(bufK, P_(EB + 13), P_(EB + 14),
                                                        P_(EB + 15), P_(EB + 16),
                                                        P_(EB + 17), P_(EB + 18),
                                                        gatesb, e, accb);
        }
    }

    // Final projection: h[B*N, 3072] @ proj_w[3072,96] + proj_b, then RevIN denorm
    mk_h<<<SZ_ / 256, 256, 0, stream>>>(bufA, bufP);
    launch_gemm(bufP, P_(165), P_(166), nullptr, 1, sbuf, B_ * N_, L_ * D_, PRED_, stream);
    finalize<<<(B_ * PRED_ * N_) / 256, 256, 0, stream>>>(sbuf, meanb, stdb, balb, (float*)d_out);
}
